// Trajectory2seq_18296560680985
// MI455X (gfx1250) — matrix operand order repaired and checked by reading, not compiled, not run on hardware
//
#include <hip/hip_runtime.h>
#include <hip/hip_bf16.h>

typedef __bf16 bf16_t;
typedef __attribute__((ext_vector_type(8)))  bf16_t v8bf;
typedef __attribute__((ext_vector_type(16))) bf16_t v16bf;
typedef __attribute__((ext_vector_type(8)))  float  v8f;

#define HID    512
#define TSTEPS 128
#define BATCH  512
#define INFEAT 457
#define DICTSZ 30
#define MAXLEN 64
#define SROW   520   // LDS A-tile row stride in bf16 elems: 512 + 8 pad (1040B = 260 dwords -> 4-bank shift/row)

// ---------------- CDNA5 async copy: global -> LDS (ASYNCcnt-tracked) ----------------
__device__ __forceinline__ void async_to_lds_b128(bf16_t* lds_dst, const bf16_t* gsrc) {
  unsigned loff = (unsigned)(uintptr_t)(__attribute__((address_space(3))) bf16_t*)lds_dst;
  asm volatile("global_load_async_to_lds_b128 %0, %1, off"
               :: "v"(loff), "v"(gsrc) : "memory");
}
__device__ __forceinline__ void wait_async_zero() {
  asm volatile("s_wait_asynccnt 0x0" ::: "memory");
}

// ---------------- WMMA fragment loaders ----------------
// A-matrix 16x32 bf16 (ISA 7.12.2): lane = half*16 + m; elements 0..7 -> K = half*8+e,
// elements 8..15 -> K = 16 + half*8 + e.  Works on global or LDS pointers.
__device__ __forceinline__ v16bf load_fragA(const bf16_t* row, int k0, int half) {
  const bf16_t* p = row + k0 + half * 8;
  v8bf lo = *(const v8bf*)(p);
  v8bf hi = *(const v8bf*)(p + 16);
  return __builtin_shufflevector(lo, hi, 0,1,2,3,4,5,6,7,8,9,10,11,12,13,14,15);
}
// B-matrix 32x16 bf16: lane = half*16 + n; elements 0..7 -> K = half*8 + e, elements 8..15 -> K = 16 + half*8 + e.
// We compute h @ W^T, so B[k][n] = W[n][k]: load row n of row-major W, two 8-value runs at k0+half*8 and +16.
__device__ __forceinline__ v16bf load_fragB(const bf16_t* row, int k0, int half) {
  const bf16_t* p = row + k0 + half * 8;
  v8bf lo = *(const v8bf*)(p);
  v8bf hi = *(const v8bf*)(p + 16);
  return __builtin_shufflevector(lo, hi, 0,1,2,3,4,5,6,7,8,9,10,11,12,13,14,15);
}
// f32-source variants with K-bound predication (for K=457 input features)
__device__ __forceinline__ v16bf load_fragA_f32(const float* row, int k0, int half, int K) {
  v16bf f;
#pragma unroll
  for (int e = 0; e < 8; ++e) {
    int ka = k0 + half * 8 + e, kb = ka + 16;
    f[e]     = (ka < K) ? (bf16_t)row[ka] : (bf16_t)0.0f;
    f[e + 8] = (kb < K) ? (bf16_t)row[kb] : (bf16_t)0.0f;
  }
  return f;
}
__device__ __forceinline__ v16bf load_fragB_f32(const float* row, int k0, int half, int K) {
  v16bf f;
#pragma unroll
  for (int e = 0; e < 16; ++e) {
    int k = k0 + half * 8 + (e < 8 ? e : e + 8);
    f[e] = (k < K) ? (bf16_t)row[k] : (bf16_t)0.0f;
  }
  return f;
}

// ---------------- unified bf16 GEMM / RNN-step kernel (K fixed at 512) ----------------
// out = [optional tanh]( A1 @ B1^T  [+ A2 @ B2^T] [+ Cbase] [+ bias] ),  N = 512.
// Block = 4 waves sharing one 16-row A tile (async-staged in LDS); wave tile M=16,N=64.
// Grid must be exact: 2 blocks per m_tile (no early exit -> EXEC all-ones for WMMA).
__global__ void __launch_bounds__(128)
gemm512_kernel(const bf16_t* __restrict__ A1, const int* __restrict__ gather1, int lda1,
               const bf16_t* __restrict__ B1,
               const bf16_t* __restrict__ A2, const bf16_t* __restrict__ B2,
               const float* __restrict__ Cbase, const float* __restrict__ bias,
               float* __restrict__ outF, bf16_t* __restrict__ out1, bf16_t* __restrict__ out2,
               int applyTanh)
{
  __shared__ __align__(16) bf16_t sA1[16 * SROW];
  __shared__ __align__(16) bf16_t sA2[16 * SROW];
  int tid  = threadIdx.x;
  int lane = tid & 31;
  int wave = blockIdx.x * 4 + (tid >> 5);
  int n_tile = wave & 7;
  int m_tile = wave >> 3;          // uniform within the block
  int m_base = m_tile * 16;
  int n_base = n_tile * 64;
  int half = lane >> 4;
  int lr   = lane & 15;

  // --- cooperative async staging of A tile(s): 16 rows x 512 bf16 = 1024 x 16B chunks ---
  for (int c = tid; c < 1024; c += 128) {
    int row = c >> 6, col = c & 63;
    int ridx = gather1 ? gather1[m_base + row] : (m_base + row);
    async_to_lds_b128(&sA1[row * SROW + col * 8], A1 + (size_t)ridx * lda1 + col * 8);
  }
  if (A2) {
    for (int c = tid; c < 1024; c += 128) {
      int row = c >> 6, col = c & 63;
      async_to_lds_b128(&sA2[row * SROW + col * 8], A2 + (size_t)(m_base + row) * HID + col * 8);
    }
  }
  wait_async_zero();
  __syncthreads();

  v8f acc[4];
#pragma unroll
  for (int j = 0; j < 4; ++j) acc[j] = (v8f){0.f,0.f,0.f,0.f,0.f,0.f,0.f,0.f};

  const bf16_t* sArow = &sA1[lr * SROW];
  for (int k0 = 0; k0 < HID; k0 += 32) {
    v16bf a = load_fragA(sArow, k0, half);            // ds_load_b128 x2
#pragma unroll
    for (int j = 0; j < 4; ++j) {
      const bf16_t* brow = B1 + (size_t)(n_base + j * 16 + lr) * HID;
      v16bf b = load_fragB(brow, k0, half);           // global, L2-resident weights
      acc[j] = __builtin_amdgcn_wmma_f32_16x16x32_bf16(false, a, false, b,
                                                       (short)0, acc[j], false, false);
    }
  }
  if (A2) {
    const bf16_t* sArow2 = &sA2[lr * SROW];
    for (int k0 = 0; k0 < HID; k0 += 32) {
      v16bf a = load_fragA(sArow2, k0, half);
#pragma unroll
      for (int j = 0; j < 4; ++j) {
        const bf16_t* brow = B2 + (size_t)(n_base + j * 16 + lr) * HID;
        v16bf b = load_fragB(brow, k0, half);
        acc[j] = __builtin_amdgcn_wmma_f32_16x16x32_bf16(false, a, false, b,
                                                         (short)0, acc[j], false, false);
      }
    }
  }
  // epilogue: C/D layout -> element v: m = m_base + half*8 + v, n = n_base + j*16 + lr
#pragma unroll
  for (int j = 0; j < 4; ++j) {
    int n = n_base + j * 16 + lr;
    float bv = bias ? bias[n] : 0.0f;
#pragma unroll
    for (int v = 0; v < 8; ++v) {
      int m = m_base + half * 8 + v;
      size_t oi = (size_t)m * HID + n;
      float val = acc[j][v] + bv;
      if (Cbase) val += Cbase[oi];
      if (applyTanh) val = tanhf(val);
      if (outF) outF[oi] = val;
      if (out1) out1[oi] = (bf16_t)val;
      if (out2) out2[oi] = (bf16_t)val;
    }
  }
}

// ---------------- XW0 precompute: [T*B,457]f32 @ [457,512] -> f32 (+bias) ----------------
__global__ void __launch_bounds__(128)
gemm_xw0_kernel(const float* __restrict__ x, const float* __restrict__ W,
                const float* __restrict__ bias, float* __restrict__ outF)
{
  int lane = threadIdx.x & 31;
  int wave = blockIdx.x * (blockDim.x >> 5) + (threadIdx.x >> 5);
  int n_tile = wave & 7;
  int m_tile = wave >> 3;                 // 0..4095
  int half = lane >> 4;
  int lr   = lane & 15;
  int r = m_tile * 16 + lr;               // row = t*512 + b
  int t = r >> 9, b = r & 511;
  const float* arow = x + ((size_t)b * TSTEPS + t) * INFEAT;
  int n_base = n_tile * 64;

  v8f acc[4];
#pragma unroll
  for (int j = 0; j < 4; ++j) acc[j] = (v8f){0.f,0.f,0.f,0.f,0.f,0.f,0.f,0.f};

  for (int k0 = 0; k0 < INFEAT; k0 += 32) {
    v16bf a = load_fragA_f32(arow, k0, half, INFEAT);
#pragma unroll
    for (int j = 0; j < 4; ++j) {
      const float* brow = W + (size_t)(n_base + j * 16 + lr) * INFEAT;
      v16bf bb = load_fragB_f32(brow, k0, half, INFEAT);
      acc[j] = __builtin_amdgcn_wmma_f32_16x16x32_bf16(false, a, false, bb,
                                                       (short)0, acc[j], false, false);
    }
  }
#pragma unroll
  for (int j = 0; j < 4; ++j) {
    int n = n_base + j * 16 + lr;
    float bv = bias[n];
#pragma unroll
    for (int v = 0; v < 8; ++v) {
      int rr = m_tile * 16 + half * 8 + v;
      outF[(size_t)rr * HID + n] = acc[j][v] + bv;
    }
  }
}

// ---------------- fc: [512,512]bf16 @ [32,512]bf16^T + b -> logits + d_out slice ----------------
__global__ void __launch_bounds__(128)
fc_kernel(const bf16_t* __restrict__ h2, const bf16_t* __restrict__ Wfc,
          const float* __restrict__ bfc, float* __restrict__ logits,
          float* __restrict__ vout, int t)
{
  int lane = threadIdx.x & 31;
  int wave = blockIdx.x * (blockDim.x >> 5) + (threadIdx.x >> 5); // 64 waves
  int n_tile = wave & 1;
  int m_tile = wave >> 1;
  int half = lane >> 4;
  int lr   = lane & 15;
  int m_base = m_tile * 16;
  int n_base = n_tile * 16;

  const bf16_t* arow = h2 + (size_t)(m_base + lr) * HID;
  const bf16_t* brow = Wfc + (size_t)(n_base + lr) * HID;
  v8f acc = (v8f){0.f,0.f,0.f,0.f,0.f,0.f,0.f,0.f};
  for (int k0 = 0; k0 < HID; k0 += 32) {
    v16bf a = load_fragA(arow, k0, half);
    v16bf b = load_fragB(brow, k0, half);
    acc = __builtin_amdgcn_wmma_f32_16x16x32_bf16(false, a, false, b,
                                                  (short)0, acc, false, false);
  }
  int n = n_base + lr;
  float bv = bfc[n];
#pragma unroll
  for (int v = 0; v < 8; ++v) {
    int m = m_base + half * 8 + v;
    float val = acc[v] + bv;
    logits[m * 32 + n] = val;
    if (n < DICTSZ) vout[(size_t)m * (DICTSZ * MAXLEN) + n * MAXLEN + t] = val;
  }
}

// ---------------- small helper kernels ----------------
__global__ void argmax_kernel(const float* __restrict__ logits, int* __restrict__ tok) {
  int b = blockIdx.x * blockDim.x + threadIdx.x;
  if (b >= BATCH) return;
  const float* l = logits + b * 32;
  float best = l[0]; int bi = 0;
  for (int n = 1; n < DICTSZ; ++n) { float v = l[n]; if (v > best) { best = v; bi = n; } }
  tok[b] = bi;
}

__global__ void cvt_bf16_kernel(const float* __restrict__ in, bf16_t* __restrict__ out, int n) {
  int i = blockIdx.x * blockDim.x + threadIdx.x;
  if (i < n) out[i] = (bf16_t)in[i];
}

__global__ void cvt_fcw_kernel(const float* __restrict__ in, bf16_t* __restrict__ out) {
  int i = blockIdx.x * blockDim.x + threadIdx.x;       // 32*512
  if (i < 32 * HID) out[i] = (i < DICTSZ * HID) ? (bf16_t)in[i] : (bf16_t)0.0f;
}

__global__ void zero_bf16_kernel(bf16_t* __restrict__ p, int n) {
  int i = blockIdx.x * blockDim.x + threadIdx.x;
  if (i < n) p[i] = (bf16_t)0.0f;
}

__global__ void prep_kernel(const float* bih0, const float* bhh0,
                            const float* bih1, const float* bhh1,
                            const float* dbih0, const float* dbhh0,
                            const float* dbih1, const float* dbhh1,
                            const float* fc_b,
                            float* b0c, float* b1c, float* d0c, float* d1c,
                            float* fcb, int* tok)
{
  int i = threadIdx.x; // 512
  b0c[i] = bih0[i] + bhh0[i];
  b1c[i] = bih1[i] + bhh1[i];
  d0c[i] = dbih0[i] + dbhh0[i];
  d1c[i] = dbih1[i] + dbhh1[i];
  if (i < 32) fcb[i] = (i < DICTSZ) ? fc_b[i] : 0.0f;
  tok[i] = 0;
}

// ---------------- host orchestration ----------------
extern "C" void kernel_launch(void* const* d_in, const int* in_sizes, int n_in,
                              void* d_out, int out_size, void* d_ws, size_t ws_size,
                              hipStream_t stream) {
  const float* x        = (const float*)d_in[0];
  const float* emb      = (const float*)d_in[1];
  const float* eWih0    = (const float*)d_in[2];
  const float* eWhh0    = (const float*)d_in[3];
  const float* ebih0    = (const float*)d_in[4];
  const float* ebhh0    = (const float*)d_in[5];
  const float* eWih1    = (const float*)d_in[6];
  const float* eWhh1    = (const float*)d_in[7];
  const float* ebih1    = (const float*)d_in[8];
  const float* ebhh1    = (const float*)d_in[9];
  const float* dWih0    = (const float*)d_in[10];
  const float* dWhh0    = (const float*)d_in[11];
  const float* dbih0    = (const float*)d_in[12];
  const float* dbhh0    = (const float*)d_in[13];
  const float* dWih1    = (const float*)d_in[14];
  const float* dWhh1    = (const float*)d_in[15];
  const float* dbih1    = (const float*)d_in[16];
  const float* dbhh1    = (const float*)d_in[17];
  const float* fc_W     = (const float*)d_in[18];
  const float* fc_b     = (const float*)d_in[19];
  float* out = (float*)d_out;

  // workspace carve-out (256B aligned)
  char* w = (char*)d_ws;
  auto alloc = [&](size_t bytes) -> void* {
    void* p = (void*)w; w += (bytes + 255) & ~(size_t)255; return p;
  };
  const size_t MAT = (size_t)HID * HID;                 // 262144 elems
  float*  XW     = (float*)alloc((size_t)TSTEPS * MAT * 4);   // reused for XW0 then XW1
  bf16_t* y0     = (bf16_t*)alloc((size_t)TSTEPS * MAT * 2);
  bf16_t* Whh0b  = (bf16_t*)alloc(MAT * 2);
  bf16_t* Wih1b  = (bf16_t*)alloc(MAT * 2);
  bf16_t* Whh1b  = (bf16_t*)alloc(MAT * 2);
  bf16_t* dWih0b = (bf16_t*)alloc(MAT * 2);
  bf16_t* dWhh0b = (bf16_t*)alloc(MAT * 2);
  bf16_t* dWih1b = (bf16_t*)alloc(MAT * 2);
  bf16_t* dWhh1b = (bf16_t*)alloc(MAT * 2);
  bf16_t* embB   = (bf16_t*)alloc((size_t)DICTSZ * HID * 2);
  bf16_t* fcWb   = (bf16_t*)alloc((size_t)32 * HID * 2);
  bf16_t* hA[2]  = { (bf16_t*)alloc(MAT * 2), (bf16_t*)alloc(MAT * 2) };
  bf16_t* hB[2]  = { (bf16_t*)alloc(MAT * 2), (bf16_t*)alloc(MAT * 2) };
  float*  b0c    = (float*)alloc(HID * 4);
  float*  b1c    = (float*)alloc(HID * 4);
  float*  d0c    = (float*)alloc(HID * 4);
  float*  d1c    = (float*)alloc(HID * 4);
  float*  fcb    = (float*)alloc(32 * 4);
  float*  logits = (float*)alloc((size_t)BATCH * 32 * 4);
  int*    tok    = (int*)alloc(BATCH * 4);
  (void)ws_size; (void)in_sizes; (void)n_in; (void)out_size;

  // 1. preprocessing
  int nM = (int)MAT;
  cvt_bf16_kernel<<<(nM + 255) / 256, 256, 0, stream>>>(eWhh0, Whh0b, nM);
  cvt_bf16_kernel<<<(nM + 255) / 256, 256, 0, stream>>>(eWih1, Wih1b, nM);
  cvt_bf16_kernel<<<(nM + 255) / 256, 256, 0, stream>>>(eWhh1, Whh1b, nM);
  cvt_bf16_kernel<<<(nM + 255) / 256, 256, 0, stream>>>(dWih0, dWih0b, nM);
  cvt_bf16_kernel<<<(nM + 255) / 256, 256, 0, stream>>>(dWhh0, dWhh0b, nM);
  cvt_bf16_kernel<<<(nM + 255) / 256, 256, 0, stream>>>(dWih1, dWih1b, nM);
  cvt_bf16_kernel<<<(nM + 255) / 256, 256, 0, stream>>>(dWhh1, dWhh1b, nM);
  cvt_bf16_kernel<<<(DICTSZ * HID + 255) / 256, 256, 0, stream>>>(emb, embB, DICTSZ * HID);
  cvt_fcw_kernel<<<(32 * HID + 255) / 256, 256, 0, stream>>>(fc_W, fcWb);
  prep_kernel<<<1, 512, 0, stream>>>(ebih0, ebhh0, ebih1, ebhh1, dbih0, dbhh0, dbih1, dbhh1,
                                     fc_b, b0c, b1c, d0c, d1c, fcb, tok);
  zero_bf16_kernel<<<(nM + 255) / 256, 256, 0, stream>>>(hA[0], nM);
  zero_bf16_kernel<<<(nM + 255) / 256, 256, 0, stream>>>(hB[0], nM);

  // 2. XW0[t,b,:] = x[b,t,:] @ Wih0^T + (bih0+bhh0)   (parallel over all T)
  gemm_xw0_kernel<<<8192, 128, 0, stream>>>(x, eWih0, b0c, XW);

  // 3. encoder layer 0 (sequential): h = tanh(XW0[t] + h @ Whh0^T); also emit y0[t] (bf16)
  for (int t = 0; t < TSTEPS; ++t) {
    int p = t & 1;
    gemm512_kernel<<<64, 128, 0, stream>>>(
        hA[p], nullptr, HID, Whh0b,
        nullptr, nullptr,
        XW + (size_t)t * MAT, nullptr,
        nullptr, hA[p ^ 1], y0 + (size_t)t * MAT, 1);
  }
  // 4. XW1 = y0 @ Wih1^T + (bih1+bhh1)  (parallel over all T*B rows, overwrites XW)
  gemm512_kernel<<<8192, 128, 0, stream>>>(
      y0, nullptr, HID, Wih1b,
      nullptr, nullptr,
      nullptr, b1c,
      XW, nullptr, nullptr, 0);

  // 5. encoder layer 1 (sequential)
  for (int t = 0; t < TSTEPS; ++t) {
    int p = t & 1;
    gemm512_kernel<<<64, 128, 0, stream>>>(
        hB[p], nullptr, HID, Whh1b,
        nullptr, nullptr,
        XW + (size_t)t * MAT, nullptr,
        nullptr, hB[p ^ 1], nullptr, 1);
  }
  // after 128 steps, final encoder hiddens live in hA[0] / hB[0] -> decoder init states

  // 6. decoder greedy loop
  float* hid0out = out + (size_t)BATCH * DICTSZ * MAXLEN;            // hidden[0]
  float* hid1out = hid0out + MAT;                                    // hidden[1]
  for (int t = 0; t < MAXLEN; ++t) {
    int p = t & 1;
    // h1 = tanh(emb[tok] @ dWih0^T + h1 @ dWhh0^T + d0c)
    gemm512_kernel<<<64, 128, 0, stream>>>(
        embB, tok, HID, dWih0b,
        hA[p], dWhh0b,
        nullptr, d0c,
        (t == MAXLEN - 1) ? hid0out : nullptr, hA[p ^ 1], nullptr, 1);
    // h2 = tanh(h1 @ dWih1^T + h2 @ dWhh1^T + d1c)
    gemm512_kernel<<<64, 128, 0, stream>>>(
        hA[p ^ 1], nullptr, HID, dWih1b,
        hB[p], dWhh1b,
        nullptr, d1c,
        (t == MAXLEN - 1) ? hid1out : nullptr, hB[p ^ 1], nullptr, 1);
    // logits + write vec_out[:, :, t]
    fc_kernel<<<16, 128, 0, stream>>>(hB[p ^ 1], fcWb, fcb, logits, out, t);
    argmax_kernel<<<2, 256, 0, stream>>>(logits, tok);
  }
}
